// MobileSelfAttention3D_60928406061686
// MI455X (gfx1250) — matrix operand order repaired and checked by reading, not compiled, not run on hardware
//
#include <hip/hip_runtime.h>

// Problem constants (B=1)
#define CCH   64
#define NC8   8
#define DDIM  16
#define HDIM  24
#define WDIM  24
#define HW    576          // HDIM*WDIM
#define NTOT  9216         // DDIM*HDIM*WDIM

typedef __attribute__((ext_vector_type(16))) _Float16 v16h;
typedef __attribute__((ext_vector_type(8)))  _Float16 v8h;
typedef __attribute__((ext_vector_type(2)))  _Float16 v2h;
typedef __attribute__((ext_vector_type(8)))  float    v8f;

union HFrag { v16h v; v8h h[2]; };

// ---------------------------------------------------------------------------
// 1) x (f32, [c][n]) -> xT (f16, [n][c])  : A-side operand of the value conv
// ---------------------------------------------------------------------------
__global__ void k_xcast(const float* __restrict__ x, _Float16* __restrict__ xT) {
  int e = blockIdx.x * blockDim.x + threadIdx.x;
  if (e < CCH * NTOT) {
    int c = e / NTOT;
    int n = e - c * NTOT;
    xT[n * CCH + c] = (_Float16)x[e];
  }
}

// ---------------------------------------------------------------------------
// 2) Pooled means + axial queries (pooling commutes with the 1x1x1 conv).
// ---------------------------------------------------------------------------
__global__ void k_prep(const float* __restrict__ x,
                       const float* __restrict__ dq_w, const float* __restrict__ dq_b,
                       const float* __restrict__ hq_w, const float* __restrict__ hq_b,
                       float* __restrict__ xd,  // [CCH][16]
                       float* __restrict__ xh,  // [CCH][24]
                       float* __restrict__ xw,  // [CCH][24]
                       float* __restrict__ dq,  // [8][16]
                       float* __restrict__ hq,  // [8][24]
                       float* __restrict__ wq)  // [8][24]
{
  const int tid = threadIdx.x;
  for (int id = tid; id < 1024 + 1536 + 1536; id += 256) {
    if (id < 1024) {                      // xd[c][l] = mean over (m,n)
      int c = id >> 4, l = id & 15;
      const float* p = x + c * NTOT + l * HW;
      float s = 0.f;
      for (int j = 0; j < HW; ++j) s += p[j];
      xd[c * DDIM + l] = s * (1.0f / HW);
    } else if (id < 2560) {               // xh[c][m] = mean over (l,n)
      int e = id - 1024;
      int c = e / HDIM, m = e - c * HDIM;
      float s = 0.f;
      for (int l = 0; l < DDIM; ++l)
        for (int n = 0; n < WDIM; ++n)
          s += x[c * NTOT + l * HW + m * WDIM + n];
      xh[c * HDIM + m] = s * (1.0f / (DDIM * WDIM));
    } else {                              // xw[c][n] = mean over (l,m)
      int e = id - 2560;
      int c = e / WDIM, n = e - c * WDIM;
      float s = 0.f;
      for (int l = 0; l < DDIM; ++l)
        for (int m = 0; m < HDIM; ++m)
          s += x[c * NTOT + l * HW + m * WDIM + n];
      xw[c * WDIM + n] = s * (1.0f / (DDIM * HDIM));
    }
  }
  __syncthreads();
  for (int id = tid; id < 128 + 192 + 192; id += 256) {
    if (id < 128) {
      int c8 = id >> 4, l = id & 15;
      float s = dq_b[c8];
      for (int c = 0; c < CCH; ++c) s += dq_w[c8 * CCH + c] * xd[c * DDIM + l];
      dq[c8 * DDIM + l] = s;
    } else if (id < 320) {
      int e = id - 128;
      int c8 = e / HDIM, m = e - c8 * HDIM;
      float s = hq_b[c8];
      for (int c = 0; c < CCH; ++c) s += hq_w[c8 * CCH + c] * xh[c * HDIM + m];
      hq[c8 * HDIM + m] = s;
    } else {                               // reference reuses hq_w/hq_b for W-query
      int e = id - 320;
      int c8 = e / WDIM, n = e - c8 * WDIM;
      float s = hq_b[c8];
      for (int c = 0; c < CCH; ++c) s += hq_w[c8 * CCH + c] * xw[c * WDIM + n];
      wq[c8 * WDIM + n] = s;
    }
  }
}

// ---------------------------------------------------------------------------
// 3) key conv: key_raw[c8][n] (f32).  The later (N,C8) view is a raw alias.
// ---------------------------------------------------------------------------
__global__ void k_key(const float* __restrict__ x,
                      const float* __restrict__ key_w, const float* __restrict__ key_b,
                      float* __restrict__ key_raw) {
  int t = blockIdx.x * blockDim.x + threadIdx.x;
  if (t < NC8 * NTOT) {
    int c8 = t / NTOT;
    int n  = t - c8 * NTOT;
    float s = key_b[c8];
    for (int c = 0; c < CCH; ++c) s += key_w[c8 * CCH + c] * x[c * NTOT + n];
    key_raw[t] = s;
  }
}

// ---------------------------------------------------------------------------
// 4) attention: per query q, key row = key_raw.flat[q*8 + c2]  (alias!)
// ---------------------------------------------------------------------------
__global__ void k_att(const float* __restrict__ key_raw,
                      const float* __restrict__ dq, const float* __restrict__ hq,
                      const float* __restrict__ wq,
                      float* __restrict__ dAtt,   // [N][16]
                      float* __restrict__ hAtt,   // [N][24]
                      float* __restrict__ wAtt)   // [N][24]
{
  int q = blockIdx.x * blockDim.x + threadIdx.x;
  if (q >= NTOT) return;
  float k8[NC8];
  for (int c2 = 0; c2 < NC8; ++c2) k8[c2] = key_raw[q * NC8 + c2];

  {
    float lg[DDIM]; float mx = -1e30f;
    for (int l = 0; l < DDIM; ++l) {
      float s = 0.f;
      for (int c2 = 0; c2 < NC8; ++c2) s += k8[c2] * dq[c2 * DDIM + l];
      lg[l] = s; mx = fmaxf(mx, s);
    }
    float sm = 0.f;
    for (int l = 0; l < DDIM; ++l) { lg[l] = __expf(lg[l] - mx); sm += lg[l]; }
    float inv = 1.0f / sm;
    for (int l = 0; l < DDIM; ++l) dAtt[q * DDIM + l] = lg[l] * inv;
  }
  {
    float lg[HDIM]; float mx = -1e30f;
    for (int m = 0; m < HDIM; ++m) {
      float s = 0.f;
      for (int c2 = 0; c2 < NC8; ++c2) s += k8[c2] * hq[c2 * HDIM + m];
      lg[m] = s; mx = fmaxf(mx, s);
    }
    float sm = 0.f;
    for (int m = 0; m < HDIM; ++m) { lg[m] = __expf(lg[m] - mx); sm += lg[m]; }
    float inv = 1.0f / sm;
    for (int m = 0; m < HDIM; ++m) hAtt[q * HDIM + m] = lg[m] * inv;
  }
  {
    float lg[WDIM]; float mx = -1e30f;
    for (int n = 0; n < WDIM; ++n) {
      float s = 0.f;
      for (int c2 = 0; c2 < NC8; ++c2) s += k8[c2] * wq[c2 * WDIM + n];
      lg[n] = s; mx = fmaxf(mx, s);
    }
    float sm = 0.f;
    for (int n = 0; n < WDIM; ++n) { lg[n] = __expf(lg[n] - mx); sm += lg[n]; }
    float inv = 1.0f / sm;
    for (int n = 0; n < WDIM; ++n) wAtt[q * WDIM + n] = lg[n] * inv;
  }
}

// ---------------------------------------------------------------------------
// 5) value conv via WMMA: V[c][n] (f16) = value_w @ x + b.
//    D(16n x 16c) = A(16n x 32c') x B(32c' x 16c); K=64 -> 2 WMMA steps.
// ---------------------------------------------------------------------------
__global__ void k_value(const _Float16* __restrict__ xT,
                        const float* __restrict__ value_w,
                        const float* __restrict__ value_b,
                        _Float16* __restrict__ Vh) {
  const int n0   = blockIdx.x * 16;
  const int wave = threadIdx.x >> 5;
  const int lane = threadIdx.x & 31;
  const int c0   = wave * 16;
  const int hi   = lane >> 4;
  const int lp   = lane & 15;

  v8f acc = {};
  for (int k0 = 0; k0 < CCH; k0 += 32) {
    HFrag a, b;
    const _Float16* arow = xT + (n0 + lp) * CCH + k0;
    a.h[0] = *(const v8h*)(arow + (hi ? 16 : 0));
    a.h[1] = *(const v8h*)(arow + (hi ? 24 : 8));
    const int ccol = c0 + lp;
    const float4* wp = (const float4*)(value_w + ccol * CCH + k0 + hi * 16);
    float4 w0 = wp[0], w1 = wp[1], w2 = wp[2], w3 = wp[3];
    const float wf[16] = { w0.x, w0.y, w0.z, w0.w, w1.x, w1.y, w1.z, w1.w,
                           w2.x, w2.y, w2.z, w2.w, w3.x, w3.y, w3.z, w3.w };
    v16h bv;
    for (int i = 0; i < 16; ++i) bv[i] = (_Float16)wf[i];
    b.v = bv;
    acc = __builtin_amdgcn_wmma_f32_16x16x32_f16(false, a.v, false, b.v,
                                                 (short)0, acc, false, false);
  }
  const int ccol = c0 + lp;
  const int nb   = n0 + hi * 8;
  const float bias = value_b[ccol];
  for (int j = 0; j < 8; j += 2) {
    _Float16 p0 = (_Float16)(acc[j]     + bias);
    _Float16 p1 = (_Float16)(acc[j + 1] + bias);
    unsigned u, u1;
    __builtin_memcpy(&u,  &p0, 2);
    __builtin_memcpy(&u1, &p1, 2);
    u |= (u1 << 16);
    *(unsigned*)(Vh + ccol * NTOT + nb + j) = u;
  }
}

// ---------------------------------------------------------------------------
// 6) main contraction: out(16q x 64c) = A(16q x 9216) x Vt(9216 x 64).
//    CDNA5 has 320 KB LDS/WG: materialize the FULL 16x9216 f16 A chunk
//    (288 KB) once, then run a barrier-free K loop: per step just
//    2x ds_load_b128 (A frag) + 2x global_load_b128 (B frag) + 1 WMMA.
//    Epilogue: out = gamma * acc + x.
// ---------------------------------------------------------------------------
__global__ void k_out(const float* __restrict__ x,
                      const float* __restrict__ dAtt, const float* __restrict__ hAtt,
                      const float* __restrict__ wAtt, const _Float16* __restrict__ Vh,
                      const float* __restrict__ gamma, float* __restrict__ out) {
  extern __shared__ char smem[];
  _Float16* Afull = (_Float16*)smem;                       // 16*9216 f16 = 294912 B
  float*    sD    = (float*)(smem + 294912);               // 16*16
  float*    sH    = (float*)(smem + 294912 + 1024);        // 16*24
  float*    sW    = (float*)(smem + 294912 + 1024 + 1536); // 16*24

  const int q0  = blockIdx.x * 16;
  const int tid = threadIdx.x;

  for (int i = tid; i < 16 * DDIM; i += 128) sD[i] = dAtt[q0 * DDIM + i];
  for (int i = tid; i < 16 * HDIM; i += 128) sH[i] = hAtt[q0 * HDIM + i];
  for (int i = tid; i < 16 * WDIM; i += 128) sW[i] = wAtt[q0 * WDIM + i];
  __syncthreads();

  // Build A[r][l*576 + m*24 + nn] = d[r,l] * (h[r,m] * w[r,nn]); hoist h*w,
  // write packed f16 pairs (rm, rm+1 never cross an m boundary since rm even).
  for (int idx = tid; idx < 16 * (HW / 2); idx += 128) {
    int r  = idx / (HW / 2);
    int p  = idx - r * (HW / 2);
    int rm = 2 * p;
    int m  = rm / WDIM;
    int nn = rm - m * WDIM;
    float hv  = sH[r * HDIM + m];
    float hw0 = hv * sW[r * WDIM + nn];
    float hw1 = hv * sW[r * WDIM + nn + 1];
    _Float16* dst = Afull + r * NTOT + rm;
    for (int l = 0; l < DDIM; ++l) {
      float dv = sD[r * DDIM + l];
      v2h pr = { (_Float16)(dv * hw0), (_Float16)(dv * hw1) };
      *(v2h*)(dst + l * HW) = pr;
    }
  }
  __syncthreads();

  const int wave = tid >> 5;
  const int lane = tid & 31;
  const int c0   = wave * 16;
  const int hi   = lane >> 4;
  const int lp   = lane & 15;

  const _Float16* arow0 = Afull + lp * NTOT + (hi ? 16 : 0);
  const _Float16* arow1 = Afull + lp * NTOT + (hi ? 24 : 8);
  const _Float16* brow0 = Vh + (c0 + lp) * NTOT + hi * 16;

  v8f acc = {};
#pragma unroll 4
  for (int kt = 0; kt < NTOT / 32; ++kt) {
    const int k0 = kt * 32;
    HFrag a, b;
    a.h[0] = *(const v8h*)(arow0 + k0);
    a.h[1] = *(const v8h*)(arow1 + k0);
    b.h[0] = *(const v8h*)(brow0 + k0);
    b.h[1] = *(const v8h*)(brow0 + k0 + 8);
    __builtin_prefetch(brow0 + k0 + 256, 0, 1);   // warm next B tiles (L2)
    acc = __builtin_amdgcn_wmma_f32_16x16x32_f16(false, a.v, false, b.v,
                                                 (short)0, acc, false, false);
  }

  const float g  = gamma[0];
  const int ccol = c0 + lp;
  const int qb   = q0 + hi * 8;       // D rows: M=j (lanes<16), M=j+8 (lanes>=16)
  for (int j = 0; j < 8; j += 2) {
    float2 v;
    v.x = g * acc[j]     + x[ccol * NTOT + qb + j];
    v.y = g * acc[j + 1] + x[ccol * NTOT + qb + j + 1];
    *(float2*)(out + ccol * NTOT + qb + j) = v;
  }
}

// ---------------------------------------------------------------------------
extern "C" void kernel_launch(void* const* d_in, const int* in_sizes, int n_in,
                              void* d_out, int out_size, void* d_ws, size_t ws_size,
                              hipStream_t stream) {
  const float* x       = (const float*)d_in[0];
  const float* key_w   = (const float*)d_in[1];
  const float* key_b   = (const float*)d_in[2];
  const float* value_w = (const float*)d_in[3];
  const float* value_b = (const float*)d_in[4];
  const float* dq_w    = (const float*)d_in[5];
  const float* dq_b    = (const float*)d_in[6];
  const float* hq_w    = (const float*)d_in[7];
  const float* hq_b    = (const float*)d_in[8];
  const float* gamma   = (const float*)d_in[9];
  float* out = (float*)d_out;

  char* w = (char*)d_ws;
  _Float16* xT      = (_Float16*)(w);                 // 589824 f16
  _Float16* Vh      = (_Float16*)(w + 1179648);       // 589824 f16
  float*    key_raw = (float*)(w + 2359296);          // 73728  f32
  float*    dAtt    = (float*)(w + 2654208);          // 147456 f32
  float*    hAtt    = (float*)(w + 3244032);          // 221184 f32
  float*    wAtt    = (float*)(w + 4128768);          // 221184 f32
  float*    xd      = (float*)(w + 5013504);
  float*    xh      = (float*)(w + 5017600);
  float*    xw      = (float*)(w + 5023744);
  float*    dq      = (float*)(w + 5029888);
  float*    hq      = (float*)(w + 5030400);
  float*    wq      = (float*)(w + 5031168);

  const size_t lds_bytes = 294912 + 1024 + 1536 + 1536;  // A chunk + att tables

  k_xcast<<<(CCH * NTOT) / 256, 256, 0, stream>>>(x, xT);
  k_prep <<<1, 256, 0, stream>>>(x, dq_w, dq_b, hq_w, hq_b, xd, xh, xw, dq, hq, wq);
  k_key  <<<(NC8 * NTOT) / 256, 256, 0, stream>>>(x, key_w, key_b, key_raw);
  k_att  <<<NTOT / 256, 256, 0, stream>>>(key_raw, dq, hq, wq, dAtt, hAtt, wAtt);
  k_value<<<NTOT / 16, 128, 0, stream>>>(xT, value_w, value_b, Vh);
  k_out  <<<NTOT / 16, 128, lds_bytes, stream>>>(x, dAtt, hAtt, wAtt, Vh, gamma, out);
}
